// Attention_33792802685741
// MI455X (gfx1250) — matrix operand order repaired and checked by reading, not compiled, not run on hardware
//
#include <hip/hip_runtime.h>
#include <hip/hip_bf16.h>

// ---- types ------------------------------------------------------------
typedef __attribute__((ext_vector_type(16))) __bf16 v16bf;
typedef __attribute__((ext_vector_type(8)))  float  v8f;
typedef __attribute__((ext_vector_type(4)))  int    v4i;

union FragBf { v16bf v; v4i q[2]; };   // 16 bf16 = 32 bytes = two b128 loads

#define NUM_HEADS 12
#define DIM_HEAD  64
#define SEQ_N     1024
#define BATCH     8
#define EMB       768
#define ROWS      (BATCH * SEQ_N)      // 8192
#define ATT_SCALE 0.125f               // 64^-0.5

static __device__ __forceinline__ v8f wmma_bf16(const FragBf& a, const FragBf& b, v8f c) {
    return __builtin_amdgcn_wmma_f32_16x16x32_bf16(false, a.v, false, b.v,
                                                   (short)0, c, false, false);
}

// A-matrix 16x32 fragment: halves 0-7 at p, halves 8-15 at p+16 (two b128)
static __device__ __forceinline__ void load_fragA(FragBf& f, const __hip_bfloat16* p) {
    f.q[0] = *(const v4i*)p;
    f.q[1] = *(const v4i*)(p + 16);
}
// B-matrix 32x16 fragment: halves 0-7 at p, halves 8-15 at p+16 (two b128)
static __device__ __forceinline__ void load_fragB(FragBf& f, const __hip_bfloat16* p) {
    f.q[0] = *(const v4i*)p;
    f.q[1] = *(const v4i*)(p + 16);
}

// CDNA5 async global->LDS copy, 16B per lane, tracked by ASYNCcnt.
// LDS dest offset = low 32 bits of flat shared address.
static __device__ __forceinline__ void async_ld_b128(unsigned lds_off, const void* gaddr) {
    asm volatile("global_load_async_to_lds_b128 %0, %1, off"
                 :: "v"(lds_off), "v"(gaddr) : "memory");
}
static __device__ __forceinline__ unsigned lds_off_of(const void* p) {
    return (unsigned)(size_t)p;
}

// ---- conversion kernels ----------------------------------------------
__global__ void cvt_bf16_kernel(const float* __restrict__ in,
                                __hip_bfloat16* __restrict__ out, int n) {
    int i = blockIdx.x * blockDim.x + threadIdx.x;
    if (i < n) out[i] = __float2bfloat16(in[i]);
}

// in [R,C] fp32 -> out [C,R] bf16  (transpose so GEMM B-fragments are contiguous)
__global__ void transpose_cvt_kernel(const float* __restrict__ in,
                                     __hip_bfloat16* __restrict__ out, int R, int C) {
    int i = blockIdx.x * blockDim.x + threadIdx.x;
    if (i < R * C) {
        int r = i / C, c = i % C;
        out[(size_t)c * R + r] = __float2bfloat16(in[i]);
    }
}

// ---- generic bf16 WMMA GEMM (register double-buffered) --
__global__ __launch_bounds__(256) void gemm_bf16_kernel(
    const __hip_bfloat16* __restrict__ A,
    const __hip_bfloat16* __restrict__ Bt,
    const float* __restrict__ bias,
    int K, int Ncol, int mode,
    __hip_bfloat16* __restrict__ Qws,
    __hip_bfloat16* __restrict__ Kws,
    __hip_bfloat16* __restrict__ Vt,
    float* __restrict__ Out)
{
    const int tid  = threadIdx.x;
    const int wid  = tid >> 5;
    const int lane = tid & 31;
    const int lo16 = lane & 15;
    const int half = lane >> 4;

    const int row0 = blockIdx.x * 256 + wid * 32;   // wave owns 32 rows
    const int col0 = blockIdx.y * 64;               // block owns 64 cols

    v8f acc[2][4] = {};

    const __hip_bfloat16* arow = A  + (size_t)(row0 + lo16) * K;
    const __hip_bfloat16* bcol = Bt + (size_t)(col0 + lo16) * K;

    auto loadAB = [&](FragBf a[2], FragBf b[4], int k0) {
#pragma unroll
        for (int m = 0; m < 2; ++m)
            load_fragA(a[m], arow + (size_t)m * 16 * K + k0 + half * 8);
#pragma unroll
        for (int c = 0; c < 4; ++c)
            load_fragB(b[c], bcol + (size_t)c * 16 * K + k0 + half * 8);
    };
    auto mma = [&](FragBf a[2], FragBf b[4]) {
#pragma unroll
        for (int c = 0; c < 4; ++c)
#pragma unroll
            for (int m = 0; m < 2; ++m)
                acc[m][c] = wmma_bf16(a[m], b[c], acc[m][c]);
    };

    FragBf a0[2], b0[4], a1[2], b1[4];
    loadAB(a0, b0, 0);
    for (int k0 = 0; k0 < K; k0 += 64) {            // K % 64 == 0
        loadAB(a1, b1, k0 + 32);
        mma(a0, b0);
        if (k0 + 64 < K) loadAB(a0, b0, k0 + 64);
        mma(a1, b1);
    }

    // epilogue: C layout -> element (m = r + 8*half, n = lo16)
#pragma unroll
    for (int mt = 0; mt < 2; ++mt) {
#pragma unroll
        for (int c = 0; c < 4; ++c) {
            const int col = col0 + c * 16 + lo16;
            const float bv = bias[col];
#pragma unroll
            for (int r = 0; r < 8; ++r) {
                const int row = row0 + mt * 16 + r + half * 8;
                const float val = acc[mt][c][r] + bv;
                if (mode == 0) {
                    const int part = col / EMB;        // 0=q 1=k 2=v
                    const int e = col % EMB;
                    const int h = e >> 6, d = e & 63;
                    const int b_ = row >> 10, n = row & 1023;
                    const size_t bh = (size_t)b_ * NUM_HEADS + h;
                    const __hip_bfloat16 bb = __float2bfloat16(val);
                    if (part == 0)      Qws[(bh * SEQ_N + n) * DIM_HEAD + d] = bb;
                    else if (part == 1) Kws[(bh * SEQ_N + n) * DIM_HEAD + d] = bb;
                    else                Vt [(bh * DIM_HEAD + d) * SEQ_N + n] = bb;
                } else {
                    Out[(size_t)row * Ncol + col] = val;
                }
            }
        }
    }
}

// ---- flash attention with async-LDS double-buffered K/V tiles ----------
// Grid: (N/64, B*H). Block: 128 threads = 4 waves; wave owns 16 query rows.
// Q,K bf16 [B,H,N,D]; Vt bf16 [B,H,D,N]; O bf16 [B,N,E].
struct SmemLayout {
    __hip_bfloat16 Kt[2][64 * 64];   // [buf][key][d]   row stride 64
    __hip_bfloat16 Vt[2][64 * 64];   // [buf][d][key]   row stride 64
    __hip_bfloat16 P [4][16 * 64];   // per-wave P tile
};

__global__ __launch_bounds__(128) void attn_kernel(
    const __hip_bfloat16* __restrict__ Q,
    const __hip_bfloat16* __restrict__ K,
    const __hip_bfloat16* __restrict__ Vt,
    __hip_bfloat16* __restrict__ O)
{
    __shared__ __align__(16) SmemLayout smem;

    const int tid  = threadIdx.x;
    const int wid  = tid >> 5;
    const int lane = tid & 31;
    const int lo16 = lane & 15;
    const int half = lane >> 4;

    const int bh = blockIdx.y;               // 0..95
    const int b_ = bh / NUM_HEADS, h = bh % NUM_HEADS;
    const int qrow0 = blockIdx.x * 64 + wid * 16;
    const __hip_bfloat16* Kbase = K  + (size_t)bh * SEQ_N * DIM_HEAD;
    const __hip_bfloat16* Vbase = Vt + (size_t)bh * DIM_HEAD * SEQ_N;

    // issue async copy of one 64-key K tile + 64-row V tile into LDS buffer.
    // 128 threads x 4 chunks x 16B = 8KB per tile; 8 async instrs per wave.
    auto stage_copy = [&](int buf, int kt) {
#pragma unroll
        for (int i = 0; i < 4; ++i) {
            const int c = tid + i * 128;          // 0..511
            const int row = c >> 3, part = c & 7; // row, 16B chunk within 128B row
            async_ld_b128(lds_off_of(&smem.Kt[buf][row * 64 + part * 8]),
                          Kbase + (size_t)(kt + row) * DIM_HEAD + part * 8);
            async_ld_b128(lds_off_of(&smem.Vt[buf][row * 64 + part * 8]),
                          Vbase + (size_t)row * SEQ_N + kt + part * 8);
        }
    };

    // load Q fragments once (A-matrix 16x32, two frags cover d=0..63)
    const __hip_bfloat16* qrow =
        Q + (size_t)bh * SEQ_N * DIM_HEAD + (size_t)(qrow0 + lo16) * DIM_HEAD;
    FragBf qf[2];
#pragma unroll
    for (int j = 0; j < 2; ++j)
        load_fragA(qf[j], qrow + j * 32 + half * 8);

    float mrow[8], lrow[8];
    v8f acc[4] = {};
#pragma unroll
    for (int r = 0; r < 8; ++r) { mrow[r] = -__builtin_inff(); lrow[r] = 0.f; }

    __hip_bfloat16* pl = &smem.P[wid][0];

    stage_copy(0, 0);                         // prologue: fill buffer 0

    for (int kt = 0; kt < SEQ_N; kt += 64) {
        const int buf = (kt >> 6) & 1;
        if (kt + 64 < SEQ_N) {
            stage_copy(buf ^ 1, kt + 64);     // overlap next tile with compute
            asm volatile("s_wait_asynccnt 8" ::: "memory");   // current tile landed
        } else {
            asm volatile("s_wait_asynccnt 0" ::: "memory");
        }
        __syncthreads();                      // all waves' copies visible

        const __hip_bfloat16* Klds = &smem.Kt[buf][0];
        const __hip_bfloat16* Vlds = &smem.Vt[buf][0];

        // ---- scores: S = Q K^T from LDS ----
        v8f s[4] = {};
#pragma unroll
        for (int j = 0; j < 2; ++j) {
            FragBf kb[4];
#pragma unroll
            for (int c = 0; c < 4; ++c)
                load_fragB(kb[c], Klds + (c * 16 + lo16) * 64 + j * 32 + half * 8);
#pragma unroll
            for (int c = 0; c < 4; ++c)
                s[c] = wmma_bf16(qf[j], kb[c], s[c]);
        }

        // ---- online softmax (row m = r + 8*half lives in 16 lanes) ----
#pragma unroll
        for (int c = 0; c < 4; ++c)
#pragma unroll
            for (int r = 0; r < 8; ++r) s[c][r] *= ATT_SCALE;

#pragma unroll
        for (int r = 0; r < 8; ++r) {
            float v = fmaxf(fmaxf(s[0][r], s[1][r]), fmaxf(s[2][r], s[3][r]));
#pragma unroll
            for (int msk = 1; msk < 16; msk <<= 1)
                v = fmaxf(v, __shfl_xor(v, msk, 32));
            const float mnew  = fmaxf(mrow[r], v);
            const float alpha = __expf(mrow[r] - mnew);
            mrow[r] = mnew;
            float psum = 0.f;
#pragma unroll
            for (int c = 0; c < 4; ++c) {
                const float p = __expf(s[c][r] - mnew);
                s[c][r] = p;
                psum += p;
            }
#pragma unroll
            for (int msk = 1; msk < 16; msk <<= 1)
                psum += __shfl_xor(psum, msk, 32);
            lrow[r] = lrow[r] * alpha + psum;
#pragma unroll
            for (int dt = 0; dt < 4; ++dt) acc[dt][r] *= alpha;
        }

        // ---- stage P through LDS: C-layout -> row-major [16 x 64] ----
#pragma unroll
        for (int c = 0; c < 4; ++c)
#pragma unroll
            for (int r = 0; r < 8; ++r)
                pl[(r + half * 8) * 64 + c * 16 + lo16] = __float2bfloat16(s[c][r]);
        // same-wave RAW on LDS: DS ops stay in order within a wave

        // ---- O += P V (P A-frags from LDS, V B-frags from LDS) ----
#pragma unroll
        for (int j = 0; j < 2; ++j) {
            FragBf pa;
            load_fragA(pa, pl + (size_t)lo16 * 64 + j * 32 + half * 8);
            FragBf vb[4];
#pragma unroll
            for (int dt = 0; dt < 4; ++dt)
                load_fragB(vb[dt], Vlds + (dt * 16 + lo16) * 64 + j * 32 + half * 8);
#pragma unroll
            for (int dt = 0; dt < 4; ++dt)
                acc[dt] = wmma_bf16(pa, vb[dt], acc[dt]);
        }

        __syncthreads();   // all reads of this buffer done before it is re-filled
    }

    // ---- finalize: O = acc / l, write bf16 [B,N,E] ----
#pragma unroll
    for (int r = 0; r < 8; ++r) {
        const float inv = 1.f / lrow[r];
        const int row = qrow0 + r + half * 8;
#pragma unroll
        for (int dt = 0; dt < 4; ++dt) {
            const int col = h * DIM_HEAD + dt * 16 + lo16;
            O[((size_t)b_ * SEQ_N + row) * EMB + col] =
                __float2bfloat16(acc[dt][r] * inv);
        }
    }
}

// ---- launch -----------------------------------------------------------
extern "C" void kernel_launch(void* const* d_in, const int* in_sizes, int n_in,
                              void* d_out, int out_size, void* d_ws, size_t ws_size,
                              hipStream_t stream) {
    const float* x     = (const float*)d_in[0];   // [8,1024,768]
    const float* Wqkv  = (const float*)d_in[1];   // [768,2304]
    const float* bqkv  = (const float*)d_in[2];   // [2304]
    const float* Wproj = (const float*)d_in[3];   // [768,768]
    const float* bproj = (const float*)d_in[4];   // [768]
    float* out = (float*)d_out;

    char* ws = (char*)d_ws;
    size_t off = 0;
    auto alloc = [&](size_t bytes) -> void* {
        void* p = ws + off;
        off += (bytes + 255) & ~(size_t)255;
        return p;
    };
    __hip_bfloat16* xb     = (__hip_bfloat16*)alloc((size_t)ROWS * EMB * 2);
    __hip_bfloat16* wqkvt  = (__hip_bfloat16*)alloc((size_t)3 * EMB * EMB * 2);
    __hip_bfloat16* wprojt = (__hip_bfloat16*)alloc((size_t)EMB * EMB * 2);
    __hip_bfloat16* Qws    = (__hip_bfloat16*)alloc((size_t)ROWS * EMB * 2);
    __hip_bfloat16* Kws    = (__hip_bfloat16*)alloc((size_t)ROWS * EMB * 2);
    __hip_bfloat16* Vtws   = (__hip_bfloat16*)alloc((size_t)ROWS * EMB * 2);
    __hip_bfloat16* Ob     = (__hip_bfloat16*)alloc((size_t)ROWS * EMB * 2);

    const int nx = ROWS * EMB;
    cvt_bf16_kernel<<<(nx + 255) / 256, 256, 0, stream>>>(x, xb, nx);
    transpose_cvt_kernel<<<(EMB * 3 * EMB + 255) / 256, 256, 0, stream>>>(
        Wqkv, wqkvt, EMB, 3 * EMB);
    transpose_cvt_kernel<<<(EMB * EMB + 255) / 256, 256, 0, stream>>>(
        Wproj, wprojt, EMB, EMB);

    // QKV projection: [8192,768] x [768,2304]
    gemm_bf16_kernel<<<dim3(ROWS / 256, (3 * EMB) / 64), 256, 0, stream>>>(
        xb, wqkvt, bqkv, EMB, 3 * EMB, 0, Qws, Kws, Vtws, nullptr);

    // attention: 16 q-tiles x 96 (b,h)
    attn_kernel<<<dim3(SEQ_N / 64, BATCH * NUM_HEADS), 128, 0, stream>>>(
        Qws, Kws, Vtws, Ob);

    // output projection: [8192,768] x [768,768] -> fp32 d_out
    gemm_bf16_kernel<<<dim3(ROWS / 256, EMB / 64), 256, 0, stream>>>(
        Ob, wprojt, bproj, EMB, EMB, 1, nullptr, nullptr, nullptr, out);
}
